// OptimizedParticleNet_21973052686572
// MI455X (gfx1250) — compile-verified, matrix operand order checked by reading, not run on hardware
//
#include <hip/hip_runtime.h>
#include <hip/hip_bf16.h>

typedef __attribute__((ext_vector_type(16))) _Float16 v16h;
typedef __attribute__((ext_vector_type(8)))  _Float16 v8h;
typedef __attribute__((ext_vector_type(8)))  float    v8f;

namespace pn {

constexpr int BGc   = 1024;            // graphs
constexpr int NPG   = 64;              // nodes per graph
constexpr int KE    = 4;               // edges per node (conv1)
constexpr int FIN   = 16;
constexpr int FG    = 10;
constexpr int H1    = 160;
constexpr int HH2   = 128;
constexpr int HH3   = 96;
constexpr int CATF  = 544;
constexpr int D0    = CATF + FG;       // 554
constexpr int D0P   = 576;             // padded to mult of 32
constexpr int NODES = BGc * NPG;       // 65536
constexpr int E1    = NODES * KE;      // 262144
constexpr float EPSV  = 1e-5f;
constexpr float SLOPE = 0.01f;

__device__ __forceinline__ float leakyf(float v) { return v >= 0.f ? v : SLOPE * v; }

// ---------------------------------------------------------------------------
// WMMA GEMM:  C[M,N](f32) = act(A[M,K](f16, lda) @ WT[N,K](f16, transposed) + bias)
// K mult of 32; M mult of 128 (true for all calls). 256 threads = 8 waves,
// block tile 128x64, per-wave 32x32 via 2x2 v_wmma_f32_16x16x32_f16.
// Double-buffered LDS, branchless main loop (last K-tile peeled), one barrier
// per step, A-stream prefetch. If partOut != nullptr, also emits deterministic
// per-block column sums / sums-of-squares (post-activation) for fused BN stats.
// ---------------------------------------------------------------------------
template<bool LEAKY>
__global__ __launch_bounds__(256)
void k_gemm(const _Float16* __restrict__ A, int lda,
            const _Float16* __restrict__ WT,      // [N][K] pre-transposed f16
            const float* __restrict__ bias,
            float* __restrict__ C,
            int M, int K, int N,
            float* __restrict__ partOut)
{
    __shared__ _Float16 sA[2][128][40];  // padded rows: conflict-free, 16B-aligned
    __shared__ _Float16 sB[2][64][40];   // B stored as [n][k]
    __shared__ float    sred[8][32][4];  // per-wave column partials for stats

    const int tid  = threadIdx.x;
    const int m0   = blockIdx.x * 128;
    const int n0   = blockIdx.y * 64;
    const int wid  = tid >> 5;
    const int lane = tid & 31;
    const int wm   = (wid & 3) * 32;     // wave M offset in block tile
    const int wn   = (wid >> 2) * 32;    // wave N offset in block tile
    const int lr   = lane & 15;
    const int khi  = (lane >> 4) * 8;    // CDNA5 16-bit A/B layout: hi half-wave = K+8

    // cooperative loader indices
    const int ar = tid >> 1;             // A tile row 0..127 (2 threads/row)
    const int ap = (tid & 1) * 16;       // A col part {0,16}
    const int br = tid >> 2;             // B tile row (n) 0..63 (4 threads/row)
    const int bp = (tid & 3) * 8;        // B col part {0,8,16,24}
    const int brc = min(n0 + br, N - 1); // clamp: dead columns load duplicates

    const _Float16* Aptr = A  + (size_t)(m0 + ar) * lda + ap;
    const _Float16* Bptr = WT + (size_t)brc * K + bp;

    v8f acc00 = {}, acc01 = {}, acc10 = {}, acc11 = {};

    auto compute = [&](int c) {
        v16h a0, a1, b0, b1;
        v8h lo, hi;
        lo = *(const v8h*)&sA[c][wm + lr][khi];      hi = *(const v8h*)&sA[c][wm + lr][khi + 16];
        a0 = __builtin_shufflevector(lo, hi, 0,1,2,3,4,5,6,7,8,9,10,11,12,13,14,15);
        lo = *(const v8h*)&sA[c][wm + 16 + lr][khi]; hi = *(const v8h*)&sA[c][wm + 16 + lr][khi + 16];
        a1 = __builtin_shufflevector(lo, hi, 0,1,2,3,4,5,6,7,8,9,10,11,12,13,14,15);
        lo = *(const v8h*)&sB[c][wn + lr][khi];      hi = *(const v8h*)&sB[c][wn + lr][khi + 16];
        b0 = __builtin_shufflevector(lo, hi, 0,1,2,3,4,5,6,7,8,9,10,11,12,13,14,15);
        lo = *(const v8h*)&sB[c][wn + 16 + lr][khi]; hi = *(const v8h*)&sB[c][wn + 16 + lr][khi + 16];
        b1 = __builtin_shufflevector(lo, hi, 0,1,2,3,4,5,6,7,8,9,10,11,12,13,14,15);
        acc00 = __builtin_amdgcn_wmma_f32_16x16x32_f16(false, a0, false, b0, (short)0, acc00, false, false);
        acc01 = __builtin_amdgcn_wmma_f32_16x16x32_f16(false, a0, false, b1, (short)0, acc01, false, false);
        acc10 = __builtin_amdgcn_wmma_f32_16x16x32_f16(false, a1, false, b0, (short)0, acc10, false, false);
        acc11 = __builtin_amdgcn_wmma_f32_16x16x32_f16(false, a1, false, b1, (short)0, acc11, false, false);
    };

    // prologue: stage K-tile 0
    {
        v8h ra0 = *(const v8h*)(Aptr);
        v8h ra1 = *(const v8h*)(Aptr + 8);
        v8h rb  = *(const v8h*)(Bptr);
        *(v8h*)&sA[0][ar][ap]     = ra0;
        *(v8h*)&sA[0][ar][ap + 8] = ra1;
        *(v8h*)&sB[0][br][bp]     = rb;
    }
    __syncthreads();

    int cur = 0;
    // branchless main loop over all but the last K-tile
    for (int kb = 0; kb + 32 < K; kb += 32) {
        v8h ra0 = *(const v8h*)(Aptr + kb + 32);
        v8h ra1 = *(const v8h*)(Aptr + kb + 40);
        v8h rb  = *(const v8h*)(Bptr + kb + 32);
        __builtin_prefetch((const void*)(Aptr + kb + 64), 0, 3);  // 2 tiles ahead

        compute(cur);

        cur ^= 1;
        *(v8h*)&sA[cur][ar][ap]     = ra0;
        *(v8h*)&sA[cur][ar][ap + 8] = ra1;
        *(v8h*)&sB[cur][br][bp]     = rb;
        __syncthreads();
    }
    compute(cur);                        // peeled last K-tile

    // C/D layout: VGPR r -> M=r (lanes 0-15), M=r+8 (lanes 16-31); N = lane&15
    const int col0 = n0 + wn + lr;
    const int col1 = n0 + wn + 16 + lr;
    const float bv0 = (col0 < N) ? bias[col0] : 0.f;
    const float bv1 = (col1 < N) ? bias[col1] : 0.f;
    float s0 = 0.f, q0 = 0.f, s1 = 0.f, q1 = 0.f;
    #pragma unroll
    for (int r = 0; r < 8; ++r) {
        const int row0 = m0 + wm + khi + r;
        const int row1 = row0 + 16;
        float o;
        if (col0 < N) {
            o = acc00[r] + bv0; o = LEAKY ? leakyf(o) : o;
            C[(size_t)row0 * N + col0] = o; s0 += o; q0 += o * o;
            o = acc10[r] + bv0; o = LEAKY ? leakyf(o) : o;
            C[(size_t)row1 * N + col0] = o; s0 += o; q0 += o * o;
        }
        if (col1 < N) {
            o = acc01[r] + bv1; o = LEAKY ? leakyf(o) : o;
            C[(size_t)row0 * N + col1] = o; s1 += o; q1 += o * o;
            o = acc11[r] + bv1; o = LEAKY ? leakyf(o) : o;
            C[(size_t)row1 * N + col1] = o; s1 += o; q1 += o * o;
        }
    }

    if (partOut) {                       // uniform branch (kernel argument)
        sred[wid][lane][0] = s0; sred[wid][lane][1] = q0;
        sred[wid][lane][2] = s1; sred[wid][lane][3] = q1;
        __syncthreads();
        if (tid < 64) {
            const int wg  = tid >> 5;        // cols 0-31 vs 32-63
            const int sel = (tid >> 4) & 1;  // col0 frag vs col1 frag
            const int l2  = tid & 15;
            float ss = 0.f, sq = 0.f;
            #pragma unroll
            for (int w = 0; w < 4; ++w) {    // fixed order => deterministic
                const int wi = wg * 4 + w;
                ss += sred[wi][l2][sel * 2]     + sred[wi][l2 + 16][sel * 2];
                sq += sred[wi][l2][sel * 2 + 1] + sred[wi][l2 + 16][sel * 2 + 1];
            }
            const int col = n0 + tid;
            if (col < N) {
                partOut[(size_t)blockIdx.x * 2 * N + col]     = ss;
                partOut[(size_t)blockIdx.x * 2 * N + N + col] = sq;
            }
        }
    }
}

// ---------------------------------------------------------------------------
// BatchNorm stats finalize (+ generic partial pass for non-GEMM tensors)
// ---------------------------------------------------------------------------
__global__ __launch_bounds__(256)
void k_stats_partial(const float* __restrict__ X, int M, int N, int rowsPer,
                     float* __restrict__ part)
{
    const int blk = blockIdx.x;
    const int r0 = blk * rowsPer;
    const int r1 = min(M, r0 + rowsPer);
    for (int n = threadIdx.x; n < N; n += blockDim.x) {
        float s = 0.f, s2 = 0.f;
        for (int r = r0; r < r1; ++r) {
            float v = X[(size_t)r * N + n];
            s += v; s2 += v * v;
        }
        part[(size_t)blk * 2 * N + n]     = s;
        part[(size_t)blk * 2 * N + N + n] = s2;
    }
}

__global__ void k_stats_final(const float* __restrict__ part, int nblk, int M, int N,
                              float* __restrict__ st)
{
    int n = blockIdx.x * blockDim.x + threadIdx.x;
    if (n >= N) return;
    float s = 0.f, s2 = 0.f;
    for (int b = 0; b < nblk; ++b) {
        s  += part[(size_t)b * 2 * N + n];
        s2 += part[(size_t)b * 2 * N + N + n];
    }
    float mean = s / (float)M;
    float var  = s2 / (float)M - mean * mean;
    st[n]     = mean;
    st[N + n] = __frsqrt_rn(fmaxf(var, 0.f) + EPSV);
}

// scalar fallback (ragged N / padded ldy — used only for the head's Z)
__global__ void k_bn_f16(const float* __restrict__ Xm, int M, int N,
                         const float* __restrict__ st,
                         const float* __restrict__ g, const float* __restrict__ be,
                         _Float16* __restrict__ Y, int ldy)
{
    size_t t = (size_t)blockIdx.x * blockDim.x + threadIdx.x;
    size_t total = (size_t)M * N;
    if (t >= total) return;
    int n = (int)(t % N);
    size_t r = t / N;
    float v = (Xm[t] - st[n]) * st[N + n] * g[n] + be[n];
    Y[r * ldy + n] = (_Float16)v;
}

// vectorized BN->f16 (N % 8 == 0, ldy == N): float4-pair loads, v8h store
__global__ void k_bn_f16v(const float* __restrict__ Xm, int M, int N,
                          const float* __restrict__ st,
                          const float* __restrict__ g, const float* __restrict__ be,
                          _Float16* __restrict__ Y)
{
    size_t t = (size_t)blockIdx.x * blockDim.x + threadIdx.x;
    const int n8 = N >> 3;
    size_t total = (size_t)M * n8;
    if (t >= total) return;
    const int c = (int)(t % n8) * 8;
    const size_t r = t / n8;
    const float* xp = Xm + r * N + c;
    float4 x0 = *(const float4*)(xp);
    float4 x1 = *(const float4*)(xp + 4);
    float xv[8] = {x0.x, x0.y, x0.z, x0.w, x1.x, x1.y, x1.z, x1.w};
    v8h out;
    #pragma unroll
    for (int q = 0; q < 8; ++q) {
        float v = (xv[q] - st[c + q]) * st[N + c + q] * g[c + q] + be[c + q];
        out[q] = (_Float16)v;
    }
    *(v8h*)&Y[r * N + c] = out;
}

// bn(mean_k(msg)) + bn(shortcut) -> concat buffer (f32 + f16 views), 8-wide
__global__ void k_aggregate(const float* __restrict__ MSG, int kne, int C,
                            const float* __restrict__ stM, const float* __restrict__ gM,
                            const float* __restrict__ beM,
                            const float* __restrict__ SC,
                            const float* __restrict__ stS, const float* __restrict__ gS,
                            const float* __restrict__ beS,
                            float* __restrict__ HC, _Float16* __restrict__ HCh, int coff)
{
    size_t t = (size_t)blockIdx.x * blockDim.x + threadIdx.x;
    const int c8 = C >> 3;
    size_t total = (size_t)NODES * c8;
    if (t >= total) return;
    const int c = (int)(t % c8) * 8;
    const size_t node = t / c8;

    float acc[8] = {0.f, 0.f, 0.f, 0.f, 0.f, 0.f, 0.f, 0.f};
    for (int kk = 0; kk < kne; ++kk) {
        const float* mp = MSG + ((size_t)node * kne + kk) * C + c;
        float4 m0 = *(const float4*)(mp);
        float4 m1 = *(const float4*)(mp + 4);
        acc[0] += m0.x; acc[1] += m0.y; acc[2] += m0.z; acc[3] += m0.w;
        acc[4] += m1.x; acc[5] += m1.y; acc[6] += m1.z; acc[7] += m1.w;
    }
    const float inv = 1.f / (float)kne;
    const float* sp = SC + (size_t)node * C + c;
    float4 sc0 = *(const float4*)(sp);
    float4 sc1 = *(const float4*)(sp + 4);
    float sv[8] = {sc0.x, sc0.y, sc0.z, sc0.w, sc1.x, sc1.y, sc1.z, sc1.w};

    float   ov[8];
    v8h     oh;
    #pragma unroll
    for (int q = 0; q < 8; ++q) {
        const int cc = c + q;
        float mv = (acc[q] * inv - stM[cc]) * stM[C + cc] * gM[cc] + beM[cc];
        float ss = (sv[q]        - stS[cc]) * stS[C + cc] * gS[cc] + beS[cc];
        float o  = mv + ss;
        ov[q] = o; oh[q] = (_Float16)o;
    }
    float* hp = HC + node * CATF + coff + c;
    *(float4*)(hp)     = make_float4(ov[0], ov[1], ov[2], ov[3]);
    *(float4*)(hp + 4) = make_float4(ov[4], ov[5], ov[6], ov[7]);
    *(v8h*)&HCh[node * CATF + coff + c] = oh;
}

// ---------------------------------------------------------------------------
// graph_norm (per graph, per feature over nodes)
// ---------------------------------------------------------------------------
__global__ __launch_bounds__(256)
void k_graph_norm(const float* __restrict__ x, const float* __restrict__ w,
                  const float* __restrict__ b, const float* __restrict__ ms,
                  float* __restrict__ xg)
{
    __shared__ float sm[FIN], sr[FIN];
    const int g = blockIdx.x, tid = threadIdx.x;
    const float* xb = x + (size_t)g * NPG * FIN;
    if (tid < FIN) {
        float mu = 0.f;
        for (int n = 0; n < NPG; ++n) mu += xb[n * FIN + tid];
        mu = ms[tid] * (mu / (float)NPG);
        float var = 0.f;
        for (int n = 0; n < NPG; ++n) { float o = xb[n * FIN + tid] - mu; var += o * o; }
        var /= (float)NPG;
        sm[tid] = mu;
        sr[tid] = __frsqrt_rn(var + EPSV);
    }
    __syncthreads();
    for (int t = tid; t < NPG * FIN; t += 256) {
        int f = t % FIN;
        float o = xb[t] - sm[f];
        xg[(size_t)g * NPG * FIN + t] = w[f] * o * sr[f] + b[f];
    }
}

__global__ void k_xg_f16(const float* __restrict__ xg, _Float16* __restrict__ y)
{
    size_t t = (size_t)blockIdx.x * blockDim.x + threadIdx.x;
    size_t total = (size_t)NODES * 32;
    if (t >= total) return;
    int c = (int)(t & 31);
    size_t r = t >> 5;
    y[t] = (c < FIN) ? (_Float16)xg[r * FIN + c] : (_Float16)0.f;
}

// conv1 edge feature builder (explicit edge list), 8-wide
__global__ void k_edges1(const int* __restrict__ EIv, const float* __restrict__ xg,
                         _Float16* __restrict__ EDGEv)
{
    size_t t = (size_t)blockIdx.x * blockDim.x + threadIdx.x;
    size_t total = (size_t)E1 * 2;           // two 8-chunks of FIN=16
    if (t >= total) return;
    const int c = (int)(t & 1) * 8;
    const size_t e = t >> 1;
    const int s  = EIv[e];
    const int tg = EIv[(size_t)E1 + e];
    const float* pi = xg + (size_t)tg * FIN + c;
    const float* pj = xg + (size_t)s  * FIN + c;
    float4 i0 = *(const float4*)(pi);
    float4 i1 = *(const float4*)(pi + 4);
    float4 j0 = *(const float4*)(pj);
    float4 j1 = *(const float4*)(pj + 4);
    float xi[8] = {i0.x, i0.y, i0.z, i0.w, i1.x, i1.y, i1.z, i1.w};
    float xj[8] = {j0.x, j0.y, j0.z, j0.w, j1.x, j1.y, j1.z, j1.w};
    v8h vi, vd;
    #pragma unroll
    for (int q = 0; q < 8; ++q) { vi[q] = (_Float16)xi[q]; vd[q] = (_Float16)(xj[q] - xi[q]); }
    *(v8h*)&EDGEv[e * 32 + c]       = vi;
    *(v8h*)&EDGEv[e * 32 + FIN + c] = vd;
}

// ---------------------------------------------------------------------------
// kNN: one workgroup per graph. 64xF features + 64x64 d2 matrix in LDS.
// ---------------------------------------------------------------------------
template<int F, int KNB>
__global__ __launch_bounds__(256)
void k_knn(const float* __restrict__ HC, int coff, int* __restrict__ idx)
{
    __shared__ float sf[64][F];
    __shared__ float sd[64][64];
    const int g = blockIdx.x, tid = threadIdx.x;
    for (int t = tid; t < 64 * F; t += 256) {
        int n = t / F, c = t % F;
        sf[n][c] = HC[(size_t)(g * 64 + n) * CATF + coff + c];
    }
    __syncthreads();
    for (int p = tid; p < 64 * 64; p += 256) {
        int i = p >> 6, j = p & 63;
        float d = 0.f;
        #pragma unroll 8
        for (int c = 0; c < F; ++c) { float df = sf[i][c] - sf[j][c]; d += df * df; }
        sd[i][j] = (i == j) ? 1e9f : d;
    }
    __syncthreads();
    if (tid < 64) {
        unsigned long long used = 0ull;
        for (int kk = 0; kk < KNB; ++kk) {
            float best = 3.4e38f; int bj = 0;
            for (int j = 0; j < 64; ++j) {
                float dv = sd[tid][j];
                if ((((used >> j) & 1ull) == 0ull) && dv < best) { best = dv; bj = j; }
            }
            used |= (1ull << bj);
            idx[(size_t)(g * 64 + tid) * KNB + kk] = bj;
        }
    }
}

// knn edge feature builder, 8-wide
template<int KNB>
__global__ void k_edges_knn(const float* __restrict__ HC, int coff, int F,
                            const int* __restrict__ idx, _Float16* __restrict__ EDGEv)
{
    size_t t = (size_t)blockIdx.x * blockDim.x + threadIdx.x;
    const int f8 = F >> 3;
    size_t total = (size_t)NODES * KNB * f8;
    if (t >= total) return;
    const int c = (int)(t % f8) * 8;
    const size_t row = t / f8;
    const int kk = (int)(row % KNB);
    const size_t node = row / KNB;
    const int b = (int)(node >> 6);
    const int j = idx[node * KNB + kk];
    const float* pi = HC + node * CATF + coff + c;
    const float* pj = HC + ((size_t)(b * 64 + j)) * CATF + coff + c;
    float4 i0 = *(const float4*)(pi);
    float4 i1 = *(const float4*)(pi + 4);
    float4 j0 = *(const float4*)(pj);
    float4 j1 = *(const float4*)(pj + 4);
    float xi[8] = {i0.x, i0.y, i0.z, i0.w, i1.x, i1.y, i1.z, i1.w};
    float xj[8] = {j0.x, j0.y, j0.z, j0.w, j1.x, j1.y, j1.z, j1.w};
    v8h vi, vd;
    #pragma unroll
    for (int q = 0; q < 8; ++q) { vi[q] = (_Float16)xi[q]; vd[q] = (_Float16)(xj[q] - xi[q]); }
    *(v8h*)&EDGEv[row * (size_t)(2 * F) + c]     = vi;
    *(v8h*)&EDGEv[row * (size_t)(2 * F) + F + c] = vd;
}

// ---------------------------------------------------------------------------
// attention + softmax pooling + head tail
// ---------------------------------------------------------------------------
__global__ void k_attn_score(const float* __restrict__ AH, const float* __restrict__ Wa2,
                             const float* __restrict__ ba2, float* __restrict__ sc)
{
    int n = blockIdx.x * blockDim.x + threadIdx.x;
    if (n >= NODES) return;
    float s = ba2[0];
    #pragma unroll
    for (int j = 0; j < 32; ++j) s += AH[(size_t)n * 32 + j] * Wa2[j];
    sc[n] = s;
}

__global__ __launch_bounds__(256)
void k_pool(const float* __restrict__ HC, const float* __restrict__ sc,
            const float* __restrict__ GIv, float* __restrict__ Z)
{
    __shared__ float wv[64];
    __shared__ float mx, ssum;
    const int g = blockIdx.x, tid = threadIdx.x;
    if (tid == 0) {
        float m = -3.4e38f;
        for (int n = 0; n < 64; ++n) m = fmaxf(m, sc[g * 64 + n]);
        mx = m;
    }
    __syncthreads();
    if (tid < 64) wv[tid] = __expf(sc[g * 64 + tid] - mx);
    __syncthreads();
    if (tid == 0) {
        float s = 0.f;
        for (int n = 0; n < 64; ++n) s += wv[n];
        ssum = s;
    }
    __syncthreads();
    for (int c = tid; c < CATF; c += 256) {
        float a = 0.f;
        for (int n = 0; n < 64; ++n) a += HC[(size_t)(g * 64 + n) * CATF + c] * wv[n];
        Z[(size_t)g * D0 + c] = a / ssum;
    }
    for (int c = tid; c < FG; c += 256)
        Z[(size_t)g * D0 + CATF + c] = GIv[g * FG + c];
}

__global__ void k_head(const float* __restrict__ D2, const float* __restrict__ st,
                       const float* __restrict__ g2, const float* __restrict__ b2,
                       const float* __restrict__ Wo, const float* __restrict__ bo,
                       float* __restrict__ out)
{
    int b = blockIdx.x * blockDim.x + threadIdx.x;
    if (b >= BGc) return;
    float s0 = bo[0], s1 = bo[1], s2 = bo[2], s3 = bo[3];
    for (int j = 0; j < 80; ++j) {
        float v = (D2[(size_t)b * 80 + j] - st[j]) * st[80 + j] * g2[j] + b2[j];
        s0 += v * Wo[j * 4 + 0];
        s1 += v * Wo[j * 4 + 1];
        s2 += v * Wo[j * 4 + 2];
        s3 += v * Wo[j * 4 + 3];
    }
    out[b * 4 + 0] = s0; out[b * 4 + 1] = s1; out[b * 4 + 2] = s2; out[b * 4 + 3] = s3;
}

// f32 [K][N] -> f16 TRANSPOSED [N][Kp] with zero pad rows K..Kp-1
__global__ void k_w2ht(const float* __restrict__ W, int K, int N,
                       _Float16* __restrict__ Y, int Kp)
{
    size_t t = (size_t)blockIdx.x * blockDim.x + threadIdx.x;
    size_t total = (size_t)N * Kp;
    if (t >= total) return;
    int k = (int)(t % Kp);
    int n = (int)(t / Kp);
    Y[t] = (k < K) ? (_Float16)W[(size_t)k * N + n] : (_Float16)0.f;
}

__global__ void k_zero_h(_Float16* p, size_t n)
{
    size_t t = (size_t)blockIdx.x * blockDim.x + threadIdx.x;
    if (t < n) p[t] = (_Float16)0.f;
}

} // namespace pn

using namespace pn;

extern "C" void kernel_launch(void* const* d_in, const int* in_sizes, int n_in,
                              void* d_out, int out_size, void* d_ws, size_t ws_size,
                              hipStream_t stream)
{
    (void)in_sizes; (void)n_in; (void)out_size; (void)ws_size;

    // ---- input pointer map (top-level insertion order; params pytree sorted keys)
    const float* X    = (const float*)d_in[0];
    const int*   EIv  = (const int*)  d_in[1];
    const float* GIv  = (const float*)d_in[2];
    // d_in[3] = batch (unused)
    const float* pWa1  = (const float*)d_in[4];
    const float* pWa2  = (const float*)d_in[5];
    const float* pWd1  = (const float*)d_in[6];
    const float* pWd2  = (const float*)d_in[7];
    const float* pWo   = (const float*)d_in[8];
    const float* pba1  = (const float*)d_in[9];
    const float* pba2  = (const float*)d_in[10];
    const float* pbd1  = (const float*)d_in[11];
    const float* pbd2  = (const float*)d_in[12];
    const float* bn0_b = (const float*)d_in[13];
    const float* bn0_g = (const float*)d_in[14];
    const float* bn1_b = (const float*)d_in[15];
    const float* bn1_g = (const float*)d_in[16];
    const float* bn2_b = (const float*)d_in[17];
    const float* bn2_g = (const float*)d_in[18];
    const float* pbo   = (const float*)d_in[19];

    struct ConvP {
        const float *W1,*W2,*W3,*Ws,*b1,*b2,*b3,*be1,*be2,*be3,*bes,*bs,*g1,*g2,*g3,*gs;
    };
    auto getconv = [&](int base) {
        ConvP p;
        p.W1  = (const float*)d_in[base + 0];
        p.W2  = (const float*)d_in[base + 1];
        p.W3  = (const float*)d_in[base + 2];
        p.Ws  = (const float*)d_in[base + 3];
        p.b1  = (const float*)d_in[base + 4];
        p.b2  = (const float*)d_in[base + 5];
        p.b3  = (const float*)d_in[base + 6];
        p.be1 = (const float*)d_in[base + 7];
        p.be2 = (const float*)d_in[base + 8];
        p.be3 = (const float*)d_in[base + 9];
        p.bes = (const float*)d_in[base + 10];
        p.bs  = (const float*)d_in[base + 11];
        p.g1  = (const float*)d_in[base + 12];
        p.g2  = (const float*)d_in[base + 13];
        p.g3  = (const float*)d_in[base + 14];
        p.gs  = (const float*)d_in[base + 15];
        return p;
    };
    ConvP C1 = getconv(20), C2 = getconv(36), C3 = getconv(52), C4 = getconv(68);
    const float* gn_b  = (const float*)d_in[84];
    const float* gn_ms = (const float*)d_in[85];
    const float* gn_w  = (const float*)d_in[86];

    // ---- workspace bump allocator
    char* wsp = (char*)d_ws;
    size_t off = 0;
    auto alloc  = [&](size_t bytes) -> char* {
        char* p = wsp + off;
        off += (bytes + 255) & ~(size_t)255;
        return p;
    };
    auto allocH = [&](size_t n) { return (_Float16*)alloc(n * sizeof(_Float16)); };
    auto allocF = [&](size_t n) { return (float*)alloc(n * sizeof(float)); };

    // f16 transposed weight copies [N][Kp] (K padded to multiples of 32)
    _Float16* c1W1h = allocH(32 * 160);
    _Float16* c1W2h = allocH(160 * 160);
    _Float16* c1W3h = allocH(160 * 160);
    _Float16* c1Wsh = allocH(32 * 160);      // 16 -> 32 pad
    _Float16* c2W1h = allocH(320 * 160);
    _Float16* c2W2h = allocH(160 * 160);
    _Float16* c2W3h = allocH(160 * 160);
    _Float16* c2Wsh = allocH(160 * 160);
    _Float16* c3W1h = allocH(320 * 128);
    _Float16* c3W2h = allocH(128 * 128);
    _Float16* c3W3h = allocH(128 * 128);
    _Float16* c3Wsh = allocH(160 * 128);
    _Float16* c4W1h = allocH(256 * 96);
    _Float16* c4W2h = allocH(96 * 96);
    _Float16* c4W3h = allocH(96 * 96);
    _Float16* c4Wsh = allocH(128 * 96);
    _Float16* Wa1h  = allocH(544 * 32);
    _Float16* Wd1h  = allocH(576 * 160);     // 554 -> 576 pad
    _Float16* Wd2h  = allocH(160 * 80);

    float*    XG   = allocF((size_t)NODES * FIN);
    _Float16* XG16 = allocH((size_t)NODES * 32);
    _Float16* EDGE = allocH((size_t)E1 * 320);
    float*    HF   = allocF((size_t)E1 * H1);
    _Float16* ACT  = allocH((size_t)E1 * H1);
    float*    SCF  = allocF((size_t)NODES * H1);
    float*    HC   = allocF((size_t)NODES * CATF);
    _Float16* HCh  = allocH((size_t)NODES * CATF);
    int*      IDX  = (int*)alloc((size_t)NODES * 4 * sizeof(int));
    float*    ATTH = allocF((size_t)NODES * 32);
    float*    SCO  = allocF((size_t)NODES);
    float*    Z    = allocF((size_t)BGc * D0);
    _Float16* Z16  = allocH((size_t)BGc * D0P);
    float*    DD1  = allocF((size_t)BGc * H1);
    _Float16* DD1h = allocH((size_t)BGc * H1);
    float*    DD2  = allocF((size_t)BGc * 80);
    float*    PART = allocF((size_t)2048 * 2 * 160 + 2 * D0P);   // per-block stats partials
    float*    STA  = allocF(2 * D0P);
    float*    STB  = allocF(2 * D0P);

    auto gsz = [](size_t tot) { return (unsigned)((tot + 255) / 256); };

    auto w2h = [&](const float* src, int K, int N, _Float16* dst, int Kp) {
        k_w2ht<<<gsz((size_t)N * Kp), 256, 0, stream>>>(src, K, N, dst, Kp);
    };
    // GEMM; if st != nullptr the per-block stats partials are emitted and folded
    auto gemm = [&](bool leaky, const _Float16* A, int lda, const _Float16* WT,
                    const float* bias, float* Cd, int M, int K, int N, float* st) {
        dim3 grid((M + 127) / 128, (N + 63) / 64);
        float* part = st ? PART : (float*)nullptr;
        if (leaky) k_gemm<true ><<<grid, 256, 0, stream>>>(A, lda, WT, bias, Cd, M, K, N, part);
        else       k_gemm<false><<<grid, 256, 0, stream>>>(A, lda, WT, bias, Cd, M, K, N, part);
        if (st) {
            int nblk = (M + 127) / 128;
            k_stats_final<<<(N + 255) / 256, 256, 0, stream>>>(PART, nblk, M, N, st);
        }
    };
    auto bnh = [&](const float* Xm, int M, int N, const float* st, const float* g,
                   const float* be, _Float16* Y, int ldy) {
        if ((N & 7) == 0 && ldy == N)
            k_bn_f16v<<<gsz((size_t)M * (N >> 3)), 256, 0, stream>>>(Xm, M, N, st, g, be, Y);
        else
            k_bn_f16<<<gsz((size_t)M * N), 256, 0, stream>>>(Xm, M, N, st, g, be, Y, ldy);
    };

    // ---- weight conversion (to transposed f16)
    w2h(C1.W1, 32, 160, c1W1h, 32);   w2h(C1.W2, 160, 160, c1W2h, 160);
    w2h(C1.W3, 160, 160, c1W3h, 160); w2h(C1.Ws, 16, 160, c1Wsh, 32);
    w2h(C2.W1, 320, 160, c2W1h, 320); w2h(C2.W2, 160, 160, c2W2h, 160);
    w2h(C2.W3, 160, 160, c2W3h, 160); w2h(C2.Ws, 160, 160, c2Wsh, 160);
    w2h(C3.W1, 320, 128, c3W1h, 320); w2h(C3.W2, 128, 128, c3W2h, 128);
    w2h(C3.W3, 128, 128, c3W3h, 128); w2h(C3.Ws, 160, 128, c3Wsh, 160);
    w2h(C4.W1, 256, 96, c4W1h, 256);  w2h(C4.W2, 96, 96, c4W2h, 96);
    w2h(C4.W3, 96, 96, c4W3h, 96);    w2h(C4.Ws, 128, 96, c4Wsh, 128);
    w2h(pWa1, 544, 32, Wa1h, 544);
    w2h(pWd1, 554, 160, Wd1h, 576);
    w2h(pWd2, 160, 80, Wd2h, 160);

    // ---- graph norm + conv1 edge inputs
    k_graph_norm<<<BGc, 256, 0, stream>>>(X, gn_w, gn_b, gn_ms, XG);
    k_xg_f16<<<gsz((size_t)NODES * 32), 256, 0, stream>>>(XG, XG16);
    k_edges1<<<gsz((size_t)E1 * 2), 256, 0, stream>>>(EIv, XG, EDGE);

    // ---- shared edge-MLP stage (stats fused into each GEMM's epilogue)
    auto conv_mlp = [&](int Me, int inK, int Cc, int kne, const ConvP& P,
                        const _Float16* W1h, const _Float16* W2hp, const _Float16* W3hp,
                        const _Float16* Wsh, const _Float16* scA, int scLda, int scK,
                        int coff) {
        gemm(true, EDGE, inK, W1h, P.b1, HF, Me, inK, Cc, STA);
        bnh(HF, Me, Cc, STA, P.g1, P.be1, ACT, Cc);
        gemm(true, ACT, Cc, W2hp, P.b2, HF, Me, Cc, Cc, STA);
        bnh(HF, Me, Cc, STA, P.g2, P.be2, ACT, Cc);
        gemm(true, ACT, Cc, W3hp, P.b3, HF, Me, Cc, Cc, STA);
        gemm(false, scA, scLda, Wsh, P.bs, SCF, NODES, scK, Cc, STB);
        k_aggregate<<<gsz((size_t)NODES * (Cc >> 3)), 256, 0, stream>>>(
            HF, kne, Cc, STA, P.g3, P.be3, SCF, STB, P.gs, P.bes, HC, HCh, coff);
    };

    // conv1 (explicit edges): 32 -> 160, shortcut 16(pad32) -> 160, c-offset 0
    conv_mlp(E1, 32, H1, KE, C1, c1W1h, c1W2h, c1W3h, c1Wsh, XG16, 32, 32, 0);

    // conv2 (knn on c1): 320 -> 160, c-offset 160
    k_knn<160, 4><<<BGc, 256, 0, stream>>>(HC, 0, IDX);
    k_edges_knn<4><<<gsz((size_t)NODES * 4 * 20), 256, 0, stream>>>(HC, 0, 160, IDX, EDGE);
    conv_mlp(E1, 320, H1, 4, C2, c2W1h, c2W2h, c2W3h, c2Wsh, HCh + 0, CATF, 160, 160);

    // conv3 (knn on c2): 320 -> 128, c-offset 320
    k_knn<160, 4><<<BGc, 256, 0, stream>>>(HC, 160, IDX);
    k_edges_knn<4><<<gsz((size_t)NODES * 4 * 20), 256, 0, stream>>>(HC, 160, 160, IDX, EDGE);
    conv_mlp(E1, 320, HH2, 4, C3, c3W1h, c3W2h, c3W3h, c3Wsh, HCh + 160, CATF, 160, 320);

    // conv4 (knn on c3, k=3): 256 -> 96, c-offset 448
    k_knn<128, 3><<<BGc, 256, 0, stream>>>(HC, 320, IDX);
    k_edges_knn<3><<<gsz((size_t)NODES * 3 * 16), 256, 0, stream>>>(HC, 320, 128, IDX, EDGE);
    conv_mlp(NODES * 3, 256, HH3, 3, C4, c4W1h, c4W2h, c4W3h, c4Wsh, HCh + 320, CATF, 128, 448);

    // ---- attention pooling (no stats needed)
    gemm(true, HCh, CATF, Wa1h, pba1, ATTH, NODES, 544, 32, nullptr);
    k_attn_score<<<gsz(NODES), 256, 0, stream>>>(ATTH, pWa2, pba2, SCO);
    k_pool<<<BGc, 256, 0, stream>>>(HC, SCO, GIv, Z);

    // ---- dense head
    k_stats_partial<<<1, 256, 0, stream>>>(Z, BGc, D0, BGc, PART);
    k_stats_final<<<(D0 + 255) / 256, 256, 0, stream>>>(PART, 1, BGc, D0, STA);
    k_zero_h<<<gsz((size_t)BGc * D0P), 256, 0, stream>>>(Z16, (size_t)BGc * D0P);
    bnh(Z, BGc, D0, STA, bn0_g, bn0_b, Z16, D0P);
    gemm(true, Z16, D0P, Wd1h, pbd1, DD1, BGc, D0P, 160, STA);
    bnh(DD1, BGc, 160, STA, bn1_g, bn1_b, DD1h, 160);
    gemm(true, DD1h, 160, Wd2h, pbd2, DD2, BGc, 160, 80, STA);
    k_head<<<(BGc + 255) / 256, 256, 0, stream>>>(DD2, STA, bn2_g, bn2_b, pWo, pbo,
                                                  (float*)d_out);
}
